// GCN_18150531793495
// MI455X (gfx1250) — compile-verified, matrix operand order checked by reading, not run on hardware
//
#include <hip/hip_runtime.h>
#include <hip/hip_bf16.h>

// ---------------------------------------------------------------------------
// Types for CDNA5 WMMA (wave32): v_wmma_f32_16x16x32_bf16
// ---------------------------------------------------------------------------
typedef __attribute__((ext_vector_type(16))) __bf16 v16bf;
typedef __attribute__((ext_vector_type(2)))  __bf16 v2bf;
typedef __attribute__((ext_vector_type(8)))  float  v8f;

__device__ __forceinline__ unsigned pack_bf16x2(float lo, float hi) {
#if __has_builtin(__builtin_amdgcn_cvt_pk_bf16_f32)
    // single v_cvt_pk_bf16_f32
    v2bf r = __builtin_amdgcn_cvt_pk_bf16_f32(lo, hi);
    return __builtin_bit_cast(unsigned, r);
#else
    // round-to-nearest-even fp32 -> bf16, packed {hi,lo}
    unsigned ulo = __builtin_bit_cast(unsigned, lo);
    unsigned uhi = __builtin_bit_cast(unsigned, hi);
    ulo = ulo + 0x7FFFu + ((ulo >> 16) & 1u);
    uhi = uhi + 0x7FFFu + ((uhi >> 16) & 1u);
    return (ulo >> 16) | (uhi & 0xFFFF0000u);
#endif
}

// ---------------------------------------------------------------------------
// Tiled WMMA GEMM: C[M x NT] = A[M x K] * B[K x NT]   (A,B fp32 row-major,
// converted to bf16 during LDS staging; f32 accumulate).
// 256 threads = 8 waves. Block tile 64(M) x NT(N); wave grid 4x2, each wave
// owns NT/32 16x16 tiles.
//
// LDS tiles are stored pre-swizzled in WMMA fragment layout: lane L's 16 bf16
// operands are 32 contiguous bytes -> fragment loads are 2x ds_load_b128.
// ISA 16-bit operand layout: element (m,k) -> lane = m + 16*((k>>3)&1),
// slot = (k&7) | ((k>>4)<<3).  A k-quad (k%4==0) stays in one lane and fills
// two adjacent dwords: dw = ((k&7)>>1) + ((k>>4)<<2)  -> one ds_store_b64.
// ---------------------------------------------------------------------------
template<int NT>
__launch_bounds__(256)
__global__ void gemm_wmma_bf16(const float* __restrict__ A,
                               const float* __restrict__ B,
                               float* __restrict__ C, int M, int K) {
    constexpr int TN = NT / 32;               // 16x16 n-tiles per wave
    __shared__ __align__(32) unsigned AsF[4][32][8];        // 4 A tiles 16x32
    __shared__ __align__(32) unsigned BsF[NT / 16][32][8];  // NT/16 B tiles 32x16

    const int t    = threadIdx.x;
    const int lane = t & 31;
    const int wid  = t >> 5;
    const int wm   = wid & 3;                 // wave M index (0..3)
    const int wn   = wid >> 2;                // wave N index (0..1)
    const int half = lane >> 4;
    const int idx  = lane & 15;
    const int mBase = blockIdx.x * 64;

    v8f acc[TN] = {};

    for (int k0 = 0; k0 < K; k0 += 32) {
        // ---- stage A: 64x32 elems = 512 k-quads, 2 per thread ----
        // Rows >= M are clamped (their products only reach never-stored C rows),
        // avoiding divergent exec-masked loads.
        #pragma unroll
        for (int i = 0; i < 2; ++i) {
            int p   = t + i * 256;
            int r   = p >> 3;                 // 0..63
            int k   = (p & 7) << 2;           // k quad base 0,4,...,28
            int row = mBase + r;
            row = row < M ? row : M - 1;
            float4 v = *(const float4*)(A + (long)row * K + k0 + k);
            int lane2 = (r & 15) | (((k >> 3) & 1) << 4);
            int dw    = ((k & 7) >> 1) + ((k >> 4) << 2);
            *(uint2*)&AsF[r >> 4][lane2][dw] =
                make_uint2(pack_bf16x2(v.x, v.y), pack_bf16x2(v.z, v.w));
        }
        // ---- stage B: 32xNT elems = 8*NT k-quads, NT/32 per thread ----
        #pragma unroll
        for (int i = 0; i < NT / 32; ++i) {
            int p = t + i * 256;
            int n = p % NT;
            int k = (p / NT) << 2;            // k quad base 0,4,...,28
            const float* bp = B + (long)(k0 + k) * NT + n;
            float l0 = bp[0];
            float l1 = bp[NT];
            float l2 = bp[2 * NT];
            float l3 = bp[3 * NT];
            int lane2 = (n & 15) | (((k >> 3) & 1) << 4);
            int dw    = ((k & 7) >> 1) + ((k >> 4) << 2);
            *(uint2*)&BsF[n >> 4][lane2][dw] =
                make_uint2(pack_bf16x2(l0, l1), pack_bf16x2(l2, l3));
        }
        __syncthreads();

        // ---- fragments: one 32B contiguous read per operand tile ----
        v16bf a = *(const v16bf*)&AsF[wm][lane][0];
        #pragma unroll
        for (int j = 0; j < TN; ++j) {
            v16bf b = *(const v16bf*)&BsF[wn * TN + j][lane][0];
            acc[j] = __builtin_amdgcn_wmma_f32_16x16x32_bf16(
                false, a, false, b, (short)0, acc[j], false, false);
        }
        __syncthreads();
    }

    // C/D layout: VGPR r, lane<16 -> M=r, lane>=16 -> M=r+8; N = idx.
    // Block-uniform fast path: only the final block is row-partial.
    if (mBase + 64 <= M) {
        #pragma unroll
        for (int j = 0; j < TN; ++j) {
            const int colBase = (wn * TN + j) * 16;
            float* cp = C + (long)(mBase + wm * 16 + half * 8) * NT + colBase + idx;
            #pragma unroll
            for (int r = 0; r < 8; ++r)
                cp[(long)r * NT] = acc[j][r];
        }
    } else {
        #pragma unroll
        for (int j = 0; j < TN; ++j) {
            const int colBase = (wn * TN + j) * 16;
            #pragma unroll
            for (int r = 0; r < 8; ++r) {
                int row = mBase + wm * 16 + r + half * 8;
                if (row < M) C[(long)row * NT + colBase + idx] = acc[j][r];
            }
        }
    }
}

// ---------------------------------------------------------------------------
// Edge scatter: agg[dst[e]][f..f+3] += h[src[e]][f..f+3] * ew[e]
// One thread per (edge, 4-feature group); float4 gather + 4 native f32 atomics.
// ---------------------------------------------------------------------------
template<int LOGF>
__global__ void scatter_edges_v4(const float* __restrict__ h,
                                 const int* __restrict__ src,
                                 const int* __restrict__ dst,
                                 const float* __restrict__ ew,
                                 float* __restrict__ agg, long E) {
    constexpr int GROUPS = 1 << (LOGF - 2);   // 4-feature groups per edge
    long i = (long)blockIdx.x * blockDim.x + threadIdx.x;
    long e = i >> (LOGF - 2);
    if (e >= E) return;
    int f = (int)(i & (GROUPS - 1)) << 2;
    int s = src[e], d = dst[e];
    float w = ew[e];
    float4 v = *(const float4*)(h + ((long)s << LOGF) + f);
    float* base = agg + ((long)d << LOGF) + f;
    __hip_atomic_fetch_add(base + 0, v.x * w, __ATOMIC_RELAXED, __HIP_MEMORY_SCOPE_AGENT);
    __hip_atomic_fetch_add(base + 1, v.y * w, __ATOMIC_RELAXED, __HIP_MEMORY_SCOPE_AGENT);
    __hip_atomic_fetch_add(base + 2, v.z * w, __ATOMIC_RELAXED, __HIP_MEMORY_SCOPE_AGENT);
    __hip_atomic_fetch_add(base + 3, v.w * w, __ATOMIC_RELAXED, __HIP_MEMORY_SCOPE_AGENT);
}

// ---------------------------------------------------------------------------
// Post layer-1: out = relu(agg + b1) * (mask >= 0.5 ? 2 : 0)   (H = 128)
// ---------------------------------------------------------------------------
__global__ void post1_kernel(const float* __restrict__ agg,
                             const float* __restrict__ b1,
                             const float* __restrict__ mask,
                             float* __restrict__ out, long n) {
    long i = (long)blockIdx.x * blockDim.x + threadIdx.x;
    if (i >= n) return;
    int f = (int)(i & 127);
    float v = agg[i] + b1[f];
    v = v > 0.0f ? v : 0.0f;
    v *= (mask[i] >= 0.5f) ? 2.0f : 0.0f;
    out[i] = v;
}

// ---------------------------------------------------------------------------
// Final: out = log_softmax(agg + b2, axis=1), 64 classes, one wave32 per row.
// ---------------------------------------------------------------------------
__global__ void logsoftmax64_kernel(const float* __restrict__ agg,
                                    const float* __restrict__ b2,
                                    float* __restrict__ out, int N) {
    int row  = blockIdx.x * (blockDim.x >> 5) + (threadIdx.x >> 5);
    int lane = threadIdx.x & 31;
    if (row >= N) return;
    const float* rp = agg + (long)row * 64;
    float v0 = rp[lane]      + b2[lane];
    float v1 = rp[lane + 32] + b2[lane + 32];
    float m = fmaxf(v0, v1);
    #pragma unroll
    for (int off = 16; off > 0; off >>= 1) m = fmaxf(m, __shfl_xor(m, off, 32));
    float s = expf(v0 - m) + expf(v1 - m);
    #pragma unroll
    for (int off = 16; off > 0; off >>= 1) s += __shfl_xor(s, off, 32);
    float lse = m + logf(s);
    out[(long)row * 64 + lane]      = v0 - lse;
    out[(long)row * 64 + lane + 32] = v1 - lse;
}

// ---------------------------------------------------------------------------
// Launch
// ---------------------------------------------------------------------------
extern "C" void kernel_launch(void* const* d_in, const int* in_sizes, int n_in,
                              void* d_out, int out_size, void* d_ws, size_t ws_size,
                              hipStream_t stream) {
    const float* x     = (const float*)d_in[0];   // [N, 512]
    const float* ew    = (const float*)d_in[1];   // [E]
    const float* W1    = (const float*)d_in[2];   // [512, 128]
    const float* b1    = (const float*)d_in[3];   // [128]
    const float* W2    = (const float*)d_in[4];   // [128, 64]
    const float* b2    = (const float*)d_in[5];   // [64]
    const float* dropm = (const float*)d_in[6];   // [N, 128]
    const int*   eidx  = (const int*)d_in[7];     // [2, E]

    const int  N   = 50000;
    const int  Fin = 512;
    const int  H   = 128;
    const int  Cc  = 64;
    const long E   = (long)in_sizes[1];

    const int* srcp = eidx;
    const int* dstp = eidx + E;

    // Workspace layout (fp32): bufA [N*H], bufB [N*H], bufC [N*C]  (~64 MB)
    float* bufA = (float*)d_ws;
    float* bufB = bufA + (size_t)N * H;
    float* bufC = bufB + (size_t)N * H;

    // 1) GEMM1: bufA = x @ W1      (50000x512 @ 512x128)
    gemm_wmma_bf16<128><<<(N + 63) / 64, 256, 0, stream>>>(x, W1, bufA, N, Fin);

    // 2) agg1 = segment_sum(bufA[src] * ew, dst)
    hipMemsetAsync(bufB, 0, (size_t)N * H * sizeof(float), stream);
    {
        long tot = E << (7 - 2);              // E * 32 groups
        scatter_edges_v4<7><<<(unsigned)((tot + 255) / 256), 256, 0, stream>>>(
            bufA, srcp, dstp, ew, bufB, E);
    }

    // 3) bufA = relu(agg1 + b1) * dropout
    {
        long nh = (long)N * H;
        post1_kernel<<<(unsigned)((nh + 255) / 256), 256, 0, stream>>>(
            bufB, b1, dropm, bufA, nh);
    }

    // 4) GEMM2: bufB = bufA @ W2   (50000x128 @ 128x64)
    gemm_wmma_bf16<64><<<(N + 63) / 64, 256, 0, stream>>>(bufA, W2, bufB, N, H);

    // 5) agg2 = segment_sum(bufB[src] * ew, dst)
    hipMemsetAsync(bufC, 0, (size_t)N * Cc * sizeof(float), stream);
    {
        long tot = E << (6 - 2);              // E * 16 groups
        scatter_edges_v4<6><<<(unsigned)((tot + 255) / 256), 256, 0, stream>>>(
            bufB, srcp, dstp, ew, bufC, E);
    }

    // 6) out = log_softmax(agg2 + b2)
    logsoftmax64_kernel<<<(N + 7) / 8, 256, 0, stream>>>(bufC, b2, (float*)d_out, N);
}
